// GCONVSTACK_WORDS_60224031425325
// MI455X (gfx1250) — matrix operand order repaired and checked by reading, not compiled, not run on hardware
//
#include <hip/hip_runtime.h>
#include <hip/hip_bf16.h>

typedef __attribute__((ext_vector_type(16))) _Float16 v16h;
typedef __attribute__((ext_vector_type(8)))  float    v8f;

#define NNODES 131072
#define NEDGES 262144
#define FIN    256
#define FH     512
#define NPAIR  (NNODES / 2)
#define HID    1536

// ---------------------------------------------------------------------------
// Zero a float buffer (grid-stride).
__global__ void gcn_zero_kernel(float* __restrict__ p, long long n) {
    long long i = (long long)blockIdx.x * blockDim.x + threadIdx.x;
    long long stride = (long long)gridDim.x * blockDim.x;
    for (; i < n; i += stride) p[i] = 0.0f;
}

// ---------------------------------------------------------------------------
// f32 -> f16 weight conversion (one-time, weights are tiny and L2-resident).
__global__ void gcn_cvt_f16_kernel(const float* __restrict__ src,
                                   _Float16* __restrict__ dst, int n) {
    int i = blockIdx.x * blockDim.x + threadIdx.x;
    int stride = gridDim.x * blockDim.x;
    for (; i < n; i += stride) dst[i] = (_Float16)src[i];
}

// ---------------------------------------------------------------------------
// Edge scatter-add: agg[dst[e], :] += feat[src[e], :]. One block per edge,
// blockDim = C/4, float4 source loads, scalar f32 atomics (global_atomic_add_f32).
__global__ void gcn_scatter_kernel(const float* __restrict__ feat,
                                   const int* __restrict__ srcIdx,
                                   const int* __restrict__ dstIdx,
                                   float* __restrict__ agg, int C) {
    int e = blockIdx.x;
    int s = srcIdx[e];
    int d = dstIdx[e];
    const float4* fs = (const float4*)(feat + (long long)s * C);
    float* ad = agg + (long long)d * C;
    int t = threadIdx.x;
    float4 v = fs[t];
    atomicAdd(ad + 4 * t + 0, v.x);
    atomicAdd(ad + 4 * t + 1, v.y);
    atomicAdd(ad + 4 * t + 2, v.z);
    atomicAdd(ad + 4 * t + 3, v.w);
}

// ---------------------------------------------------------------------------
// Fused dual GEMM + bias + ReLU using v_wmma_f32_16x16x32_f16.
//   out[m, n] = relu( sum_k A1[m,k]*Wa[n,k] + sum_k A2[m,k]*Wb[n,k] + bias[n] )
// Weights are pre-converted f16 (row-major (Ncols, K)); activations are f32,
// converted in registers. A2 may be null (single GEMM). If outX1 != null
// (layer 2), rows are split even/odd into outX1/outX2 (x1/x2 views of hc).
//
// Grid: x = column tiles (fast dispatch dim), y = row strips. All column
// blocks of one row strip are adjacent in dispatch order, so the 256 KB A
// strip streams from HBM once and re-reads hit L2; weights stay L2-resident.
//
// Block: 128 threads = 4 waves. Wave w owns rows [by*128 + w*32, +32) and
// cols [bx*64, +64): 2 A fragments x 4 B fragments -> 8 WMMAs / k-step.
__global__ __launch_bounds__(128)
void gcn_wmma_gemm_kernel(const float* __restrict__ A1, const _Float16* __restrict__ Wa,
                          const float* __restrict__ A2, const _Float16* __restrict__ Wb,
                          const float* __restrict__ bias,
                          float* __restrict__ outMain,
                          float* __restrict__ outX1, float* __restrict__ outX2,
                          int K, int Ncols) {
    const int lane = threadIdx.x & 31;
    const int wave = threadIdx.x >> 5;
    const int m15  = lane & 15;       // row-in-tile (A), col-in-tile (B, C/D)
    const int hi   = lane >> 4;       // half-wave select
    const long long rowBase = (long long)blockIdx.y * 128 + wave * 32;
    const int colBase = blockIdx.x * 64;

    v8f acc[4][2] = {};

    for (int pass = 0; pass < 2; ++pass) {
        const float* A = pass ? A2 : A1;
        const _Float16* W = pass ? Wb : Wa;
        if (A == nullptr) break;
        const float* arow0 = A + (rowBase + m15) * (long long)K;
        const float* arow1 = A + (rowBase + 16 + m15) * (long long)K;
        for (int k0 = 0; k0 < K; k0 += 32) {
            // A fragments: lane holds row m15 (+16); K = hi*16 + {0..15}
            const float* ap0 = arow0 + k0 + hi * 16;
            const float* ap1 = arow1 + k0 + hi * 16;
            v16h a0, a1;
#pragma unroll
            for (int j = 0; j < 8; ++j) {
                a0[j]     = (_Float16)ap0[j];
                a0[8 + j] = (_Float16)ap0[8 + j];
                a1[j]     = (_Float16)ap1[j];
                a1[8 + j] = (_Float16)ap1[8 + j];
            }
            if (k0 + 32 < K) {
                __builtin_prefetch(ap0 + 32, 0, 1);  // global_prefetch_b8
                __builtin_prefetch(ap1 + 32, 0, 1);
            }
#pragma unroll
            for (int t = 0; t < 4; ++t) {
                // B fragment: lane holds col m15; K = hi*16 + {0..15}; direct f16 load
                const _Float16* wrow =
                    W + (long long)(colBase + t * 16 + m15) * K + k0 + hi * 16;
                v16h bf = *(const v16h*)wrow;
                acc[t][0] = __builtin_amdgcn_wmma_f32_16x16x32_f16(
                    false, a0, false, bf, (short)0, acc[t][0], false, false);
                acc[t][1] = __builtin_amdgcn_wmma_f32_16x16x32_f16(
                    false, a1, false, bf, (short)0, acc[t][1], false, false);
            }
        }
    }

    // Epilogue: bias + ReLU (+ optional even/odd row split for layer 2).
    // Streaming outputs -> non-temporal stores (don't rinse L2 weight/A strip).
#pragma unroll
    for (int t = 0; t < 4; ++t) {
        const int col = colBase + t * 16 + m15;
        const float bv = bias ? bias[col] : 0.0f;
#pragma unroll
        for (int ri = 0; ri < 2; ++ri) {
#pragma unroll
            for (int r = 0; r < 8; ++r) {
                long long row = rowBase + ri * 16 + r + hi * 8;  // C/D: M = r + hi*8
                float v = acc[t][ri][r] + bv;
                v = v > 0.0f ? v : 0.0f;
                __builtin_nontemporal_store(v, &outMain[row * Ncols + col]);
                if (outX1) {
                    float* dst = (row & 1) ? outX2 : outX1;
                    __builtin_nontemporal_store(
                        v, &dst[(row >> 1) * (long long)Ncols + col]);
                }
            }
        }
    }
}

// ---------------------------------------------------------------------------
// Head: out[i] = sigmoid( hid[i,:] . W2 + b2 ). One wave32 per row.
__global__ void gcn_head_kernel(const float* __restrict__ hid,
                                const float* __restrict__ w2,
                                const float* __restrict__ b2,
                                float* __restrict__ out, int rows) {
    int gwave = (int)((blockIdx.x * (long long)blockDim.x + threadIdx.x) >> 5);
    int lane  = threadIdx.x & 31;
    if (gwave >= rows) return;
    const float* h = hid + (long long)gwave * HID;
    float s = 0.0f;
    for (int i = lane; i < HID; i += 32) s += h[i] * w2[i];
#pragma unroll
    for (int off = 16; off > 0; off >>= 1) s += __shfl_xor(s, off, 32);
    if (lane == 0) out[gwave] = 1.0f / (1.0f + __expf(-(s + b2[0])));
}

// ---------------------------------------------------------------------------
extern "C" void kernel_launch(void* const* d_in, const int* in_sizes, int n_in,
                              void* d_out, int out_size, void* d_ws, size_t ws_size,
                              hipStream_t stream) {
    (void)in_sizes; (void)n_in; (void)out_size; (void)ws_size;

    const float* x     = (const float*)d_in[0];
    const int*   ei    = (const int*)d_in[1];      // (2, E): row 0 = src, row 1 = dst
    const float* Wrel0 = (const float*)d_in[3];
    const float* brel0 = (const float*)d_in[4];
    const float* Wroot0= (const float*)d_in[5];
    const float* Wrel1 = (const float*)d_in[6];
    const float* brel1 = (const float*)d_in[7];
    const float* Wroot1= (const float*)d_in[8];
    const float* Wrel2 = (const float*)d_in[9];
    const float* brel2 = (const float*)d_in[10];
    const float* Wroot2= (const float*)d_in[11];
    const float* W1    = (const float*)d_in[12];
    const float* b1    = (const float*)d_in[13];
    const float* W2    = (const float*)d_in[14];
    const float* b2    = (const float*)d_in[15];

    const int* srcIdx = ei;
    const int* dstIdx = ei + NEDGES;

    // Output layout: out (N/2) | hc (N*512) | x1 (N/2*512) | x2 (N/2*512)
    float* out = (float*)d_out;
    float* hc  = out + NPAIR;
    float* x1  = hc + (long long)NNODES * FH;
    float* x2  = x1 + (long long)NPAIR * FH;

    // Workspace: agg | h0 | h1 | f16 weights. hid aliases h0+h1 (dead by MLP).
    float* agg = (float*)d_ws;
    float* h0  = agg + (long long)NNODES * FH;
    float* h1  = h0  + (long long)NNODES * FH;
    float* hid = h0;   // (N/2)*1536 floats, spans dead h0+h1 region
    _Float16* wf = (_Float16*)(h1 + (long long)NNODES * FH);

    _Float16* fWrel0  = wf;                         // 512*256
    _Float16* fWroot0 = fWrel0  + FH * FIN;         // 512*256
    _Float16* fWrel1  = fWroot0 + FH * FIN;         // 512*512
    _Float16* fWroot1 = fWrel1  + FH * FH;
    _Float16* fWrel2  = fWroot1 + FH * FH;
    _Float16* fWroot2 = fWrel2  + FH * FH;
    _Float16* fW1     = fWroot2 + FH * FH;          // 1536*1024

    // ---- One-time weight conversion to f16 ----
    gcn_cvt_f16_kernel<<<512, 256, 0, stream>>>(Wrel0,  fWrel0,  FH * FIN);
    gcn_cvt_f16_kernel<<<512, 256, 0, stream>>>(Wroot0, fWroot0, FH * FIN);
    gcn_cvt_f16_kernel<<<1024, 256, 0, stream>>>(Wrel1,  fWrel1,  FH * FH);
    gcn_cvt_f16_kernel<<<1024, 256, 0, stream>>>(Wroot1, fWroot1, FH * FH);
    gcn_cvt_f16_kernel<<<1024, 256, 0, stream>>>(Wrel2,  fWrel2,  FH * FH);
    gcn_cvt_f16_kernel<<<1024, 256, 0, stream>>>(Wroot2, fWroot2, FH * FH);
    gcn_cvt_f16_kernel<<<3072, 256, 0, stream>>>(W1,     fW1,     HID * 2 * FH);

    dim3 blk(128);
    dim3 gL(FH / 64, NNODES / 128);       // x = 8 col tiles (fast), y = 1024 row strips
    dim3 gM(HID / 64, NPAIR / 128);       // x = 24 col tiles (fast), y = 512 row strips

    // ---- Layer 0: K = 256 ----
    gcn_zero_kernel<<<4096, 256, 0, stream>>>(agg, (long long)NNODES * FIN);
    gcn_scatter_kernel<<<NEDGES, FIN / 4, 0, stream>>>(x, srcIdx, dstIdx, agg, FIN);
    gcn_wmma_gemm_kernel<<<gL, blk, 0, stream>>>(agg, fWrel0, x, fWroot0, brel0,
                                                 h0, nullptr, nullptr, FIN, FH);
    // ---- Layer 1: K = 512 ----
    gcn_zero_kernel<<<4096, 256, 0, stream>>>(agg, (long long)NNODES * FH);
    gcn_scatter_kernel<<<NEDGES, FH / 4, 0, stream>>>(h0, srcIdx, dstIdx, agg, FH);
    gcn_wmma_gemm_kernel<<<gL, blk, 0, stream>>>(agg, fWrel1, h0, fWroot1, brel1,
                                                 h1, nullptr, nullptr, FH, FH);
    // ---- Layer 2: K = 512, writes hc and splits into x1/x2 ----
    gcn_zero_kernel<<<4096, 256, 0, stream>>>(agg, (long long)NNODES * FH);
    gcn_scatter_kernel<<<NEDGES, FH / 4, 0, stream>>>(h1, srcIdx, dstIdx, agg, FH);
    gcn_wmma_gemm_kernel<<<gL, blk, 0, stream>>>(agg, fWrel2, h1, fWroot2, brel2,
                                                 hc, x1, x2, FH, FH);
    // ---- MLP: hid = relu(hc @ W1^T + b1), K = 1024, N = 1536 ----
    gcn_wmma_gemm_kernel<<<gM, blk, 0, stream>>>(hc, fW1, nullptr, nullptr, b1,
                                                 hid, nullptr, nullptr, 2 * FH, HID);
    // ---- Head: sigmoid dot, one wave per row ----
    gcn_head_kernel<<<(NPAIR * 32) / 256, 256, 0, stream>>>(hid, W2, b2, out, NPAIR);
}
